// ScaledDotProductAttention_2551210574097
// MI455X (gfx1250) — compile-verified, matrix operand order checked by reading, not run on hardware
//
#include <hip/hip_runtime.h>
#include <hip/hip_bf16.h>

// ---------------------------------------------------------------------------
// CDNA5 (gfx1250) fused multi-head attention, bf16 WMMA + TDM async tensor DMA.
//   B=16, N=784, D=512, H=8, DK=DV=64
// ---------------------------------------------------------------------------

typedef __attribute__((ext_vector_type(16))) __bf16 v16bf;
typedef __attribute__((ext_vector_type(8)))  float  v8f;
typedef __attribute__((ext_vector_type(4)))  unsigned int v4u;
typedef __attribute__((ext_vector_type(8)))  int v8i;
typedef __attribute__((ext_vector_type(4)))  int v4i;

#define Bb   16
#define Nn   784
#define Dd   512
#define Hh   8
#define DK   64
#define NPAD 800            // padded key dim so 32-wide chunks are readable
#define NEG_BIG (-1.0e30f)

// ---- per-wave dynamic-LDS layout for the flash kernel (bytes) ----
#define KROW   72                     // K tile row stride in elems (64 + 4-DWORD TDM pad)
#define VROW   40                     // V tile row stride in elems (32 + 4-DWORD TDM pad)
#define KTILE_B (32 * KROW * 2)       // 4608
#define VTILE_B (64 * VROW * 2)       // 5120
#define POFF_B  (2 * KTILE_B + 2 * VTILE_B)
#define SLOT_B  (POFF_B + 16 * 40 * 2)          // 20736 (16B aligned)

// TDM group1 word0: data_size=2B (code 1), pad_enable, pad_interval, pad_amount=4 DWORDs (code 3)
#define TDM_FLAGS_K ((1u << 16) | (1u << 20) | (4u << 22) | (3u << 25))  // 32-DWORD rows
#define TDM_FLAGS_V ((1u << 16) | (1u << 20) | (3u << 22) | (3u << 25))  // 16-DWORD rows

__device__ __forceinline__ __bf16 f2bf(float f) { return (__bf16)f; }

__device__ __forceinline__ v8f wmma_bf16(v16bf a, v16bf b, v8f c) {
  return __builtin_amdgcn_wmma_f32_16x16x32_bf16(
      false, a, false, b, (short)0, c, false, false);
}

// Fragment loader for row-major bf16 matrix (global or LDS): rows row0..+15,
// cols col0..+31.  ISA 16-bit A layout: lane<16 -> M=lane, K {0..7,16..23}.
__device__ __forceinline__ v16bf load_frag_bf16(const __bf16* base,
                                                int row0, int stride, int col0) {
  int lane = threadIdx.x & 31;
  int half = lane >> 4;
  int r    = lane & 15;
  const __bf16* p = base + (size_t)(row0 + r) * stride + col0 + half * 8;
  union { v16bf v; uint4 q[2]; } u;
  u.q[0] = *(const uint4*)(p);
  u.q[1] = *(const uint4*)(p + 16);
  return u.v;
}

// Same, but f32 source converted to bf16 in-register (v_cvt_pk_bf16_f32).
__device__ __forceinline__ v16bf load_frag_f32(const float* __restrict__ base,
                                               int row0, int stride, int col0) {
  int lane = threadIdx.x & 31;
  int half = lane >> 4;
  int r    = lane & 15;
  const float* p = base + (size_t)(row0 + r) * stride + col0 + half * 8;
  float t[16];
  *(float4*)(t + 0)  = *(const float4*)(p + 0);
  *(float4*)(t + 4)  = *(const float4*)(p + 4);
  *(float4*)(t + 8)  = *(const float4*)(p + 16);
  *(float4*)(t + 12) = *(const float4*)(p + 20);
  v16bf v;
#pragma unroll
  for (int i = 0; i < 16; ++i) v[i] = f2bf(t[i]);
  return v;
}

// ---------------------------------------------------------------------------
// TDM 2D tile load: builds the D# (ISA ch.8) and issues tensor_load_to_lds.
// Tracked with TENSORcnt.  All inputs are wave-uniform.
// (6-arg clang-23 builtin: g0, g1, g2, g3, extra group, cpol)
// ---------------------------------------------------------------------------
__device__ __forceinline__ void tdm_load_2d(unsigned lds_off, const void* gptr,
                                            unsigned tile_d0, unsigned tile_d1,
                                            unsigned tens_d0, unsigned tens_d1,
                                            unsigned d0_stride, unsigned flags) {
  unsigned long long ga = (unsigned long long)(uintptr_t)gptr;
  v4u g0;
  g0[0] = 1u;                                            // count=1, user descriptor
  g0[1] = lds_off;                                       // LDS byte address
  g0[2] = (unsigned)(ga & 0xffffffffull);                // global_addr[31:0]
  g0[3] = (unsigned)((ga >> 32) & 0x1ffffffull) | (2u << 30);  // addr[56:32] | type=2
  v8i g1;
  g1[0] = (int)flags;                                    // data_size | pad ctl
  g1[1] = (int)((tens_d0 & 0xffffu) << 16);              // tensor_dim0[15:0]
  g1[2] = (int)((tens_d0 >> 16) | ((tens_d1 & 0xffffu) << 16));
  g1[3] = (int)((tens_d1 >> 16) | (tile_d0 << 16));      // | tile_dim0
  g1[4] = (int)(tile_d1 & 0xffffu);                      // tile_dim1 (tile_dim2=0)
  g1[5] = (int)d0_stride;                                // tensor_dim0_stride[31:0]
  g1[6] = 0;                                             // stride msbs / dim1_stride
  g1[7] = 0;
  v4i z4 = {0, 0, 0, 0};
  v8i z8 = {0, 0, 0, 0, 0, 0, 0, 0};
  __builtin_amdgcn_tensor_load_to_lds(g0, g1, z4, z4, z8, 0);
}

// ---------------------------------------------------------------------------
// Kernel 0: transpose + convert the 4 weight matrices (512x512 f32 -> bf16^T)
// ---------------------------------------------------------------------------
__global__ void wtrans_kernel(const float* __restrict__ W0, const float* __restrict__ W1,
                              const float* __restrict__ W2, const float* __restrict__ W3,
                              __bf16* __restrict__ T0, __bf16* __restrict__ T1,
                              __bf16* __restrict__ T2, __bf16* __restrict__ T3) {
  int idx = blockIdx.x * 256 + threadIdx.x;
  int w   = idx >> 18;
  int rem = idx & ((1 << 18) - 1);
  int k   = rem >> 9;
  int n   = rem & 511;
  const float* W = (w == 0) ? W0 : (w == 1) ? W1 : (w == 2) ? W2 : W3;
  __bf16*      T = (w == 0) ? T0 : (w == 1) ? T1 : (w == 2) ? T2 : T3;
  T[(size_t)n * 512 + k] = f2bf(W[(size_t)k * 512 + n]);
}

// ---------------------------------------------------------------------------
// Kernel 1: fused QKV projection, software-pipelined K loop.
// ---------------------------------------------------------------------------
__global__ void qkv_proj_kernel(const float* __restrict__ Xq, const float* __restrict__ Xk,
                                const float* __restrict__ Xv,
                                const __bf16* __restrict__ Wqt, const __bf16* __restrict__ Wkt,
                                const __bf16* __restrict__ Wvt,
                                const float* __restrict__ bq, const float* __restrict__ bk,
                                const float* __restrict__ bv,
                                __bf16* __restrict__ Qb, __bf16* __restrict__ Kb,
                                __bf16* __restrict__ Vt) {
  int wave  = blockIdx.x * (blockDim.x >> 5) + (threadIdx.x >> 5);
  int which = wave / (Nn / 16 * (Dd / 16));   // 0=Q 1=K 2=V
  int t     = wave % (Nn / 16 * (Dd / 16));
  int mt    = t / (Dd / 16);
  int nt    = t % (Dd / 16);

  const float*  X    = (which == 0) ? Xq  : (which == 1) ? Xk  : Xv;
  const __bf16* Wt   = (which == 0) ? Wqt : (which == 1) ? Wkt : Wvt;
  const float*  bias = (which == 0) ? bq  : (which == 1) ? bk  : bv;

  v8f acc = {};
  v16bf a_cur = load_frag_f32(X, mt * 16, Dd, 0);
  v16bf b_cur = load_frag_bf16(Wt, nt * 16, Dd, 0);
#pragma unroll 4
  for (int kb = 0; kb < Dd - 32; kb += 32) {          // pipelined: prefetch k+1
    v16bf a_nxt = load_frag_f32(X, mt * 16, Dd, kb + 32);
    v16bf b_nxt = load_frag_bf16(Wt, nt * 16, Dd, kb + 32);
    acc = wmma_bf16(a_cur, b_cur, acc);
    a_cur = a_nxt;
    b_cur = b_nxt;
  }
  acc = wmma_bf16(a_cur, b_cur, acc);

  int lane = threadIdx.x & 31, half = lane >> 4, nl = lane & 15;
  int c = nt * 16 + nl;
  int h = c >> 6;
  int d = c & 63;
  float bscal = bias[c];
#pragma unroll
  for (int i = 0; i < 8; ++i) {
    int m  = i + 8 * half;
    int g  = mt * 16 + m;
    int b_ = g / Nn;
    int nr = g % Nn;
    __bf16 v = f2bf(acc[i] + bscal);
    if (which == 0)
      Qb[((size_t)(b_ * Hh + h) * Nn + nr) * DK + d] = v;
    else if (which == 1)
      Kb[((size_t)(b_ * Hh + h) * NPAD + nr) * DK + d] = v;
    else
      Vt[((size_t)(b_ * Hh + h) * DK + d) * NPAD + nr] = v;   // transposed
  }
}

// ---------------------------------------------------------------------------
// Kernel 2: flash attention.  One wave per (b, h, 16-row q-tile).
// K / V^T chunks stream through LDS via double-buffered TDM tensor loads
// (TENSORcnt);  S/PV on bf16 WMMA;  softmax via half-wave shuffles.
// ---------------------------------------------------------------------------
__global__ void __launch_bounds__(128)
flash_attn_kernel(const __bf16* __restrict__ Qb, const __bf16* __restrict__ Kb,
                  const __bf16* __restrict__ Vt,
                  const float* __restrict__ self_corr, const float* __restrict__ aw,
                  const unsigned char* __restrict__ mask,
                  __bf16* __restrict__ Obuf) {
  extern __shared__ char smem[];                 // dynamic only -> offsets start at 0
  int wib  = threadIdx.x >> 5;
  int wave = blockIdx.x * 4 + wib;
  int qt   = wave % (Nn / 16);
  int bh   = wave / (Nn / 16);
  int h    = bh & (Hh - 1);
  int b    = bh >> 3;

  const __bf16* K = Kb + (size_t)bh * NPAD * DK;
  const __bf16* V = Vt + (size_t)bh * DK * NPAD;
  const float* scb = self_corr + (size_t)b * Nn * Nn;
  const float* awb = aw        + (size_t)b * Nn * Nn;
  const unsigned char* mkb = mask + (size_t)b * Nn * Nn;

  unsigned slot = (unsigned)(wib * SLOT_B);
  __bf16* Kl0 = (__bf16*)(smem + slot);
  __bf16* Kl1 = (__bf16*)(smem + slot + KTILE_B);
  __bf16* Vl0 = (__bf16*)(smem + slot + 2 * KTILE_B);
  __bf16* Vl1 = (__bf16*)(smem + slot + 2 * KTILE_B + VTILE_B);
  __bf16* P   = (__bf16*)(smem + slot + POFF_B);

  int lane = threadIdx.x & 31, half = lane >> 4, nl = lane & 15;
  int q0 = qt * 16;

  v16bf qf0 = load_frag_bf16(Qb + (size_t)bh * Nn * DK, q0, DK, 0);
  v16bf qf1 = load_frag_bf16(Qb + (size_t)bh * Nn * DK, q0, DK, 32);

  v8f acc[4] = {{}, {}, {}, {}};
  float rowmax[8], rowsum[8];
#pragma unroll
  for (int i = 0; i < 8; ++i) { rowmax[i] = NEG_BIG; rowsum[i] = 0.f; }

  // ---- issue TDM for chunk 0 (K: 32x64 contiguous;  V^T: 64x32 strided) ----
  tdm_load_2d(slot, K, 64, 32, 64, NPAD, 64, TDM_FLAGS_K);
  tdm_load_2d(slot + 2 * KTILE_B, V, 32, 64, NPAD, 64, NPAD, TDM_FLAGS_V);

  for (int kc = 0; kc < 25; ++kc) {
    int kbase = kc * 32;
    int cur   = kc & 1;
    __bf16* Kl = cur ? Kl1 : Kl0;
    __bf16* Vl = cur ? Vl1 : Vl0;

    // ---- prefetch chunk kc+1 into the other buffer (fully async) ----
    if (kc + 1 < 25) {
      int nb = kbase + 32;
      unsigned kboff = slot + (cur ? 0 : KTILE_B);             // opposite buffer
      unsigned vboff = slot + 2 * KTILE_B + (cur ? 0 : VTILE_B);
      tdm_load_2d(kboff, K + (size_t)nb * DK, 64, 32, 64,
                  (unsigned)(NPAD - nb), 64, TDM_FLAGS_K);
      tdm_load_2d(vboff, V + nb, 32, 64, (unsigned)(NPAD - nb), 64,
                  NPAD, TDM_FLAGS_V);
      __builtin_amdgcn_s_wait_tensorcnt(2);   // chunk kc resident; kc+1 in flight
    } else {
      __builtin_amdgcn_s_wait_tensorcnt(0);
    }
    asm volatile("" ::: "memory");

    // ---- S = Q K^T from LDS K tile (rows=keys, stride KROW elems) ----
    v8f s0 = {}, s1 = {};
    {
      v16bf kf = load_frag_bf16(Kl, 0, KROW, 0);
      s0 = wmma_bf16(qf0, kf, s0);
      kf = load_frag_bf16(Kl, 0, KROW, 32);
      s0 = wmma_bf16(qf1, kf, s0);
      kf = load_frag_bf16(Kl, 16, KROW, 0);
      s1 = wmma_bf16(qf0, kf, s1);
      kf = load_frag_bf16(Kl, 16, KROW, 32);
      s1 = wmma_bf16(qf1, kf, s1);
    }

    // ---- fused epilogue: scale, +corr, *weight, mask ----
    float x0[8], x1[8];
    int k0 = kbase + nl;
    int k1 = kbase + 16 + nl;
    bool v1 = (k1 < Nn);
#pragma unroll
    for (int i = 0; i < 8; ++i) {
      int m = i + 8 * half;
      size_t i0 = (size_t)(q0 + m) * Nn + k0;
      float a0 = (s0[i] * 0.125f + scb[i0]) * awb[i0];
      x0[i] = mkb[i0] ? NEG_BIG : a0;
      if (v1) {
        size_t i1 = i0 + 16;
        float a1 = (s1[i] * 0.125f + scb[i1]) * awb[i1];
        x1[i] = mkb[i1] ? NEG_BIG : a1;
      } else {
        x1[i] = NEG_BIG;
      }
    }

    // ---- online softmax (row's 16 columns live in one half-wave) ----
    float ml[8];
#pragma unroll
    for (int i = 0; i < 8; ++i) ml[i] = fmaxf(x0[i], x1[i]);
#pragma unroll
    for (int off = 1; off <= 8; off <<= 1)
#pragma unroll
      for (int i = 0; i < 8; ++i) ml[i] = fmaxf(ml[i], __shfl_xor(ml[i], off, 32));

    float alpha[8], ps[8];
#pragma unroll
    for (int i = 0; i < 8; ++i) {
      float nm = fmaxf(rowmax[i], ml[i]);
      alpha[i]  = __expf(rowmax[i] - nm);
      rowmax[i] = nm;
      x0[i] = __expf(x0[i] - nm);
      x1[i] = __expf(x1[i] - nm);
      ps[i] = x0[i] + x1[i];
    }
#pragma unroll
    for (int off = 1; off <= 8; off <<= 1)
#pragma unroll
      for (int i = 0; i < 8; ++i) ps[i] += __shfl_xor(ps[i], off, 32);
#pragma unroll
    for (int i = 0; i < 8; ++i) rowsum[i] = rowsum[i] * alpha[i] + ps[i];
#pragma unroll
    for (int j = 0; j < 4; ++j)
#pragma unroll
      for (int i = 0; i < 8; ++i) acc[j][i] *= alpha[i];

    // ---- P (C-layout f32) -> LDS bf16 -> A-fragment ----
#pragma unroll
    for (int i = 0; i < 8; ++i) {
      int m = i + 8 * half;
      P[m * 40 + nl]      = f2bf(x0[i]);
      P[m * 40 + 16 + nl] = f2bf(x1[i]);
    }
    asm volatile("s_wait_dscnt 0x0" ::: "memory");

    v16bf pf;
    {
      const __bf16* pp = P + (size_t)nl * 40 + half * 8;
      union { v16bf v; uint4 q[2]; } u;
      u.q[0] = *(const uint4*)(pp);
      u.q[1] = *(const uint4*)(pp + 16);
      pf = u.v;
    }
    asm volatile("s_wait_dscnt 0x0" ::: "memory");

    // ---- O += P * V from LDS V^T tile (rows=d, stride VROW elems) ----
#pragma unroll
    for (int j = 0; j < 4; ++j) {
      v16bf vf = load_frag_bf16(Vl, j * 16, VROW, 0);
      acc[j] = wmma_bf16(pf, vf, acc[j]);
    }
  }

  // ---- normalize and store O (bf16, (b, n, h*64+d) layout) ----
  float inv[8];
#pragma unroll
  for (int i = 0; i < 8; ++i) inv[i] = 1.0f / rowsum[i];
#pragma unroll
  for (int j = 0; j < 4; ++j)
#pragma unroll
    for (int i = 0; i < 8; ++i) {
      int m = i + 8 * half;
      Obuf[((size_t)b * Nn + (q0 + m)) * Dd + h * 64 + j * 16 + nl] =
          f2bf(acc[j][i] * inv[i]);
    }
}

// ---------------------------------------------------------------------------
// Kernel 3: output projection + fused (b,n,c) -> (b,c,28,28) transpose store.
// ---------------------------------------------------------------------------
__global__ void oproj_kernel(const __bf16* __restrict__ Obuf, const __bf16* __restrict__ Wot,
                             const float* __restrict__ bo, float* __restrict__ out) {
  int wave = blockIdx.x * (blockDim.x >> 5) + (threadIdx.x >> 5);
  int mt = wave / (Dd / 16);
  int nt = wave % (Dd / 16);

  v8f acc = {};
  v16bf a_cur = load_frag_bf16(Obuf, mt * 16, Dd, 0);
  v16bf b_cur = load_frag_bf16(Wot,  nt * 16, Dd, 0);
#pragma unroll 4
  for (int kb = 0; kb < Dd - 32; kb += 32) {
    v16bf a_nxt = load_frag_bf16(Obuf, mt * 16, Dd, kb + 32);
    v16bf b_nxt = load_frag_bf16(Wot,  nt * 16, Dd, kb + 32);
    acc = wmma_bf16(a_cur, b_cur, acc);
    a_cur = a_nxt;
    b_cur = b_nxt;
  }
  acc = wmma_bf16(a_cur, b_cur, acc);

  int lane = threadIdx.x & 31, half = lane >> 4, nl = lane & 15;
  int c = nt * 16 + nl;
  float bias = bo[c];
#pragma unroll
  for (int i = 0; i < 8; ++i) {
    int m  = i + 8 * half;
    int g  = mt * 16 + m;
    int b_ = g / Nn;
    int q  = g % Nn;
    out[((size_t)b_ * Dd + c) * Nn + q] = acc[i] + bias;
  }
}

// ---------------------------------------------------------------------------
// Host-side launcher (graph-capture safe: kernels + hipMemsetAsync only)
// ---------------------------------------------------------------------------
extern "C" void kernel_launch(void* const* d_in, const int* in_sizes, int n_in,
                              void* d_out, int out_size, void* d_ws, size_t ws_size,
                              hipStream_t stream) {
  const float* queries = (const float*)d_in[0];
  const float* keys    = (const float*)d_in[1];
  const float* values  = (const float*)d_in[2];
  const float* sc      = (const float*)d_in[3];
  const float* aw      = (const float*)d_in[4];
  const unsigned char* mk = (const unsigned char*)d_in[5];
  const float* Wq = (const float*)d_in[6];
  const float* bq = (const float*)d_in[7];
  const float* Wk = (const float*)d_in[8];
  const float* bk = (const float*)d_in[9];
  const float* Wv = (const float*)d_in[10];
  const float* bv = (const float*)d_in[11];
  const float* Wo = (const float*)d_in[12];
  const float* bo = (const float*)d_in[13];
  float* out = (float*)d_out;

  char* ws = (char*)d_ws;
  size_t off = 0;
  __bf16* Qb  = (__bf16*)(ws + off); off += (size_t)Bb * Hh * Nn   * DK * 2;
  __bf16* Kb  = (__bf16*)(ws + off); off += (size_t)Bb * Hh * NPAD * DK * 2;
  __bf16* Vt  = (__bf16*)(ws + off); off += (size_t)Bb * Hh * DK * NPAD * 2;
  __bf16* Ob  = (__bf16*)(ws + off); off += (size_t)Bb * Nn * Dd * 2;
  __bf16* Wqt = (__bf16*)(ws + off); off += (size_t)Dd * Dd * 2;
  __bf16* Wkt = (__bf16*)(ws + off); off += (size_t)Dd * Dd * 2;
  __bf16* Wvt = (__bf16*)(ws + off); off += (size_t)Dd * Dd * 2;
  __bf16* Wot = (__bf16*)(ws + off); off += (size_t)Dd * Dd * 2;

  // clear pad rows/cols of K/V (poisoned workspace must not leak into tiles)
  (void)hipMemsetAsync(d_ws, 0, off, stream);

  wtrans_kernel<<<dim3(4096), dim3(256), 0, stream>>>(Wq, Wk, Wv, Wo, Wqt, Wkt, Wvt, Wot);

  qkv_proj_kernel<<<dim3(3 * (Nn / 16) * (Dd / 16) / 4), dim3(128), 0, stream>>>(
      queries, keys, values, Wqt, Wkt, Wvt, bq, bk, bv, Qb, Kb, Vt);

  flash_attn_kernel<<<dim3(Bb * Hh * (Nn / 16) / 4), dim3(128), 4 * SLOT_B, stream>>>(
      Qb, Kb, Vt, sc, aw, mk, Ob);

  oproj_kernel<<<dim3((Nn / 16) * (Dd / 16) / 4), dim3(128), 0, stream>>>(Ob, Wot, bo, out);
}
